// SRULayer_48223892799512
// MI455X (gfx1250) — compile-verified, matrix operand order checked by reading, not run on hardware
//
#include <hip/hip_runtime.h>
#include <hip/hip_bf16.h>
#include <math.h>

// ---------------------------------------------------------------------------
// SRU layer, MI455X (gfx1250, wave32):
//   k0: x f32 -> bf16            (one-time, enables async byte-copy staging)
//   k1: W f32 -> bf16 transposed ([N][K] so B tiles are K-contiguous)
//   k2: u = x@W + b via bf16 WMMA 16x16x32, async global->LDS double buffer,
//       hardware-tanh sigmoid epilogue, scatter to 3 planes
//   k3: chunked scan pass 1 (per-chunk affine summary A,B)
//   k4: chunked scan pass 2 (compose prefixes, emit h with v_tanh_f32)
// ---------------------------------------------------------------------------

typedef __attribute__((ext_vector_type(16))) __bf16 v16bf;
typedef __attribute__((ext_vector_type(8)))  __bf16 v8bf;
typedef __attribute__((ext_vector_type(4)))  __bf16 v4bf;
typedef __attribute__((ext_vector_type(8)))  float  v8f;
typedef __attribute__((ext_vector_type(4)))  float  v4f;
typedef __attribute__((ext_vector_type(4)))  int    v4i;

static constexpr int Bb = 16, Tt = 2048, Dd = 1024, Hh = 1024;
static constexpr int Mm = Bb * Tt;      // 32768
static constexpr int Nn = 3 * Hh;       // 3072
static constexpr int NC = 16;           // scan chunks per sequence
static constexpr int CL = Tt / NC;      // 128 steps per chunk

#define BM 128
#define BN 128
#define BK 32
#define KPAD 40   // halves: 80B rows -> 16B aligned, 20-bank stride (conflict-free)

#if defined(__gfx1250__) && __has_builtin(__builtin_amdgcn_global_load_async_to_lds_b128)
#define HAVE_ASYNC 1
#else
#define HAVE_ASYNC 0
#endif

__device__ __forceinline__ __bf16 f2bf(float f) {
    union { float f; unsigned u; } v; v.f = f;
    unsigned r = v.u + 0x7FFFu + ((v.u >> 16) & 1u);   // RNE
    unsigned short hv = (unsigned short)(r >> 16);
    __bf16 out;
    __builtin_memcpy(&out, &hv, 2);
    return out;
}

__device__ __forceinline__ float fast_tanh(float x) {
#if __has_builtin(__builtin_amdgcn_tanhf)
    return __builtin_amdgcn_tanhf(x);
#elif __has_builtin(__builtin_amdgcn_tanh_f32)
    return __builtin_amdgcn_tanh_f32(x);
#else
    float r; asm("v_tanh_f32 %0, %1" : "=v"(r) : "v"(x)); return r;
#endif
}

__device__ __forceinline__ float fast_sigmoid(float x) {
    return fmaf(fast_tanh(0.5f * x), 0.5f, 0.5f);
}

#if HAVE_ASYNC
typedef __attribute__((address_space(1))) v4i* gptr4;   // <4 x i32> addrspace(1)*
typedef __attribute__((address_space(3))) v4i* lptr4;   // <4 x i32> addrspace(3)*

__device__ __forceinline__ void async_cp16(const __bf16* g, __bf16* l) {
    __builtin_amdgcn_global_load_async_to_lds_b128((gptr4)g, (lptr4)l, 0, 0);
}
template <int N>
__device__ __forceinline__ void wait_async() {
    __builtin_amdgcn_s_wait_asynccnt(N);
}
#endif

// ---- k0: convert x to bf16 -------------------------------------------------
__global__ __launch_bounds__(256)
void conv_x(const float* __restrict__ X, __bf16* __restrict__ Xb) {
    size_t i = ((size_t)blockIdx.x * 256 + threadIdx.x) * 4;
    v4f x = *(const v4f*)(X + i);
    v4bf h;
    h[0] = f2bf(x[0]); h[1] = f2bf(x[1]); h[2] = f2bf(x[2]); h[3] = f2bf(x[3]);
    *(v4bf*)(Xb + i) = h;
}

// ---- k1: convert + transpose W -> Wt[N][K] bf16 ----------------------------
__global__ __launch_bounds__(256)
void conv_wt(const float* __restrict__ W, __bf16* __restrict__ Wt) {
    int gid = blockIdx.x * 256 + threadIdx.x;   // Nn * (Dd/4) threads
    int n = gid >> 8;                           // Dd/4 == 256
    int k = (gid & 255) * 4;
    v4bf h;
    #pragma unroll
    for (int i = 0; i < 4; ++i) h[i] = f2bf(W[(size_t)(k + i) * Nn + n]);
    *(v4bf*)(Wt + (size_t)n * Dd + k) = h;
}

// ---- k2: GEMM + activation -------------------------------------------------
__global__ __launch_bounds__(256)
void sru_gemm_act(const __bf16* __restrict__ Xb, const __bf16* __restrict__ Wt,
                  const float* __restrict__ bias,
                  float* __restrict__ xtb, float* __restrict__ fbuf,
                  float* __restrict__ rbuf)
{
    __shared__ __bf16 AsB[2][BM * KPAD];
    __shared__ __bf16 BsB[2][BM * KPAD];

    const int tid  = threadIdx.x;
    const int lane = tid & 31;
    const int wave = tid >> 5;
    const int wm   = (wave & 3) * 32;
    const int wn   = (wave >> 2) * 64;
    const int bn0  = blockIdx.x * BN;
    const int bm0  = blockIdx.y * BM;

    v8f acc[2][4];
    #pragma unroll
    for (int i = 0; i < 2; ++i)
        #pragma unroll
        for (int j = 0; j < 4; ++j)
            #pragma unroll
            for (int e = 0; e < 8; ++e)
                acc[i][j][e] = 0.0f;

    // tile staging: 512 16B-chunks per tile (128 rows x 4), 2 per thread
    const int c0 = tid * 2;
    const int fr = lane & 15;          // fragment row (M for A, N for B)
    const int kh = (lane >> 4) * 8;    // ISA K-half select

    auto compute = [&](int bi) {
        v16bf a[2], bm_[4];
        #pragma unroll
        for (int i = 0; i < 2; ++i) {
            const __bf16* p = &AsB[bi][(wm + i * 16 + fr) * KPAD + kh];
            v8bf lo = *(const v8bf*)(p);
            v8bf hi = *(const v8bf*)(p + 16);
            #pragma unroll
            for (int q = 0; q < 8; ++q) { a[i][q] = lo[q]; a[i][q + 8] = hi[q]; }
        }
        #pragma unroll
        for (int j = 0; j < 4; ++j) {
            const __bf16* p = &BsB[bi][(wn + j * 16 + fr) * KPAD + kh];
            v8bf lo = *(const v8bf*)(p);
            v8bf hi = *(const v8bf*)(p + 16);
            #pragma unroll
            for (int q = 0; q < 8; ++q) { bm_[j][q] = lo[q]; bm_[j][q + 8] = hi[q]; }
        }
        #pragma unroll
        for (int i = 0; i < 2; ++i)
            #pragma unroll
            for (int j = 0; j < 4; ++j)
                acc[i][j] = __builtin_amdgcn_wmma_f32_16x16x32_bf16(
                    false, a[i], false, bm_[j], (short)0, acc[i][j], false, false);
    };

#if HAVE_ASYNC
    auto stage = [&](int bi, int k0) {
        #pragma unroll
        for (int u = 0; u < 2; ++u) {
            int c = c0 + u;
            int row = c >> 2, off = (c & 3) * 8;
            async_cp16(Xb + (size_t)(bm0 + row) * Dd + k0 + off,
                       &AsB[bi][row * KPAD + off]);
            async_cp16(Wt + (size_t)(bn0 + row) * Dd + k0 + off,
                       &BsB[bi][row * KPAD + off]);
        }
    };
    stage(0, 0);
    for (int kt = 0; kt < Dd / BK; ++kt) {
        if (kt + 1 < Dd / BK) {
            stage((kt + 1) & 1, (kt + 1) * BK);
            wait_async<4>();                      // current tile's 4 ops done
        } else {
            wait_async<0>();
        }
        __syncthreads();
        compute(kt & 1);
        __syncthreads();                          // readers done before overwrite
    }
#else
    for (int kt = 0; kt < Dd / BK; ++kt) {
        int k0 = kt * BK;
        v8bf ra[2], rb[2];
        #pragma unroll
        for (int u = 0; u < 2; ++u) {
            int c = c0 + u;
            int row = c >> 2, off = (c & 3) * 8;
            ra[u] = *(const v8bf*)(Xb + (size_t)(bm0 + row) * Dd + k0 + off);
            rb[u] = *(const v8bf*)(Wt + (size_t)(bn0 + row) * Dd + k0 + off);
        }
        __syncthreads();
        #pragma unroll
        for (int u = 0; u < 2; ++u) {
            int c = c0 + u;
            int row = c >> 2, off = (c & 3) * 8;
            *(v8bf*)(&AsB[0][row * KPAD + off]) = ra[u];
            *(v8bf*)(&BsB[0][row * KPAD + off]) = rb[u];
        }
        __syncthreads();
        compute(0);
    }
#endif

    // epilogue: each 128-col block lies wholly in one third (1024 % 128 == 0)
    const int part = bn0 / Hh;
    float* dst = (part == 0) ? xtb : (part == 1) ? fbuf : rbuf;
    const int cn0 = bn0 - part * Hh;
    const int mo  = (lane >> 4) * 8;
    #pragma unroll
    for (int i = 0; i < 2; ++i) {
        #pragma unroll
        for (int j = 0; j < 4; ++j) {
            int gn = bn0 + wn + j * 16 + fr;
            int cn = cn0 + wn + j * 16 + fr;
            float bv = bias[gn];
            #pragma unroll
            for (int e = 0; e < 8; ++e) {
                int gm = bm0 + wm + i * 16 + mo + e;
                float v = acc[i][j][e] + bv;
                if (part != 0) v = fast_sigmoid(v);
                dst[(size_t)gm * Hh + cn] = v;
            }
        }
    }
}

// ---- k3: scan pass 1 — per-chunk affine summary (A = prod f, B = local scan)
__global__ __launch_bounds__(256)
void scan_pass1(const float* __restrict__ xtb, const float* __restrict__ fbuf,
                float* __restrict__ Ac, float* __restrict__ Bc)
{
    int gid = blockIdx.x * 256 + threadIdx.x;   // (b*NC + chunk)*H + h
    int h  = gid & (Hh - 1);
    int bc = gid >> 10;
    int b = bc >> 4, chunk = bc & (NC - 1);
    size_t base = ((size_t)b * Tt + (size_t)chunk * CL) * Hh + h;
    float a = 1.0f, cb = 0.0f;
    #pragma unroll 8
    for (int t = 0; t < CL; ++t) {
        float ft = fbuf[base];
        float xt = xtb[base];
        a  *= ft;
        cb  = ft * cb + (1.0f - ft) * xt;
        base += (size_t)Hh;
    }
    Ac[gid] = a;
    Bc[gid] = cb;
}

// ---- k4: scan pass 2 — compose chunk prefix, emit h ------------------------
__global__ __launch_bounds__(256)
void scan_pass2(const float* __restrict__ xtb, const float* __restrict__ fbuf,
                const float* __restrict__ rbuf, const float* __restrict__ Ac,
                const float* __restrict__ Bc, float* __restrict__ out)
{
    int gid = blockIdx.x * 256 + threadIdx.x;
    int h  = gid & (Hh - 1);
    int bc = gid >> 10;
    int b = bc >> 4, chunk = bc & (NC - 1);

    float c = 0.0f;                              // compose prefixes (<=15 steps)
    size_t cbase = (size_t)(b * NC) * Hh + h;
    for (int i = 0; i < chunk; ++i)
        c = Ac[cbase + (size_t)i * Hh] * c + Bc[cbase + (size_t)i * Hh];

    size_t base = ((size_t)b * Tt + (size_t)chunk * CL) * Hh + h;
    #pragma unroll 4
    for (int t = 0; t < CL; ++t) {
        float ft = fbuf[base];
        float xt = xtb[base];
        float rt = rbuf[base];
        c = ft * c + (1.0f - ft) * xt;
        out[base] = rt * fast_tanh(c) + (1.0f - rt) * xt;
        base += (size_t)Hh;
    }
}

extern "C" void kernel_launch(void* const* d_in, const int* in_sizes, int n_in,
                              void* d_out, int out_size, void* d_ws, size_t ws_size,
                              hipStream_t stream)
{
    const float* X    = (const float*)d_in[0];   // [B, T, D]
    const float* W    = (const float*)d_in[1];   // [D, 3H]
    const float* bias = (const float*)d_in[2];   // [3H]
    float* out = (float*)d_out;                  // [B, T, H]

    char* w = (char*)d_ws;
    float*  xtb  = (float*)w;  w += (size_t)Mm * Hh * 4;
    float*  fbuf = (float*)w;  w += (size_t)Mm * Hh * 4;
    float*  rbuf = (float*)w;  w += (size_t)Mm * Hh * 4;
    __bf16* Xb   = (__bf16*)w; w += (size_t)Mm * Dd * 2;
    __bf16* Wt   = (__bf16*)w; w += (size_t)Nn * Dd * 2;
    float*  Ac   = (float*)w;  w += (size_t)Bb * NC * Hh * 4;
    float*  Bc   = (float*)w;

    conv_x <<<(Mm * (size_t)Dd / 4) / 256, 256, 0, stream>>>(X, Xb);
    conv_wt<<<(Nn * (size_t)Dd / 4) / 256, 256, 0, stream>>>(W, Wt);

    dim3 grid(Nn / BN, Mm / BM);                 // 24 x 256
    sru_gemm_act<<<grid, 256, 0, stream>>>(Xb, Wt, bias, xtb, fbuf, rbuf);

    int scan_threads = Bb * NC * Hh;             // 262144
    scan_pass1<<<scan_threads / 256, 256, 0, stream>>>(xtb, fbuf, Ac, Bc);
    scan_pass2<<<scan_threads / 256, 256, 0, stream>>>(xtb, fbuf, rbuf, Ac, Bc, out);
}
